// CGNN_75118978007103
// MI455X (gfx1250) — matrix operand order repaired and checked by reading, not compiled, not run on hardware
//
#include <hip/hip_runtime.h>
#include <hip/hip_bf16.h>

// ---------------- problem constants (from reference) ----------------
#define NN    50000
#define EE    800000
#define F_IN  92
#define FF    128
#define DD    50
#define LL    3
#define GG    256

typedef __attribute__((ext_vector_type(16))) __bf16 v16bf;
typedef __attribute__((ext_vector_type(8)))  float  v8f;

union ABu {
    v16bf          v;
    uint4          q[2];
    unsigned short h[16];
};

__device__ __forceinline__ unsigned short f2bf_bits(float f) {
    unsigned int u = __float_as_uint(f);
    unsigned int r = (u + 0x7FFFu + ((u >> 16) & 1u)) >> 16;   // RNE
    return (unsigned short)r;
}

__device__ __forceinline__ v8f wmma_bf16(const ABu& a, const ABu& b, v8f c) {
    return __builtin_amdgcn_wmma_f32_16x16x32_bf16(
        /*neg_a=*/false, a.v, /*neg_b=*/false, b.v,
        /*c_mod=*/(short)0, c, /*reuse_a=*/false, /*reuse_b=*/false);
}

// ---------------- utility kernels ----------------
__global__ void zero_u32(unsigned int* __restrict__ p, long long n) {
    long long i = (long long)blockIdx.x * blockDim.x + threadIdx.x;
    if (i < n) p[i] = 0u;
}

// f32 [M,K] row-major -> bf16 [M,Kp] row-major, zero padded
__global__ void cvt_pad_bf16(const float* __restrict__ X, int K, int Kp,
                             unsigned short* __restrict__ O, long long total) {
    long long i = (long long)blockIdx.x * blockDim.x + threadIdx.x;
    if (i >= total) return;
    int k = (int)(i % Kp);
    long long m = i / Kp;
    O[i] = (k < K) ? f2bf_bits(X[m * K + k]) : (unsigned short)0;
}

// Pack f32 weight submatrix [Ksrc(+pad), 128] into the WMMA B-fragment layout.
// Output u32 index: (((nt*KT + ktBegin+ktl)*32 + lane)*8 + i)
// holds bf16 pair (K = ktl*32 + ((lane>>4)<<3) + 2i + (i>=4 ? 8 : 0), +1 ; N = nt*16 + (lane&15))
__global__ void packB(const float* __restrict__ W, int Ksrc,
                      int ktBegin, int ktCount, int KT,
                      unsigned int* __restrict__ out) {
    int idx = blockIdx.x * blockDim.x + threadIdx.x;
    int total = 8 * ktCount * 32 * 8;
    if (idx >= total) return;
    int i    = idx & 7;
    int lane = (idx >> 3) & 31;
    int ktl  = (idx >> 8) % ktCount;
    int nt   = (idx >> 8) / ktCount;
    int row0 = ktl * 32 + ((lane >> 4) << 3) + 2 * i + (i >= 4 ? 8 : 0);
    int col  = nt * 16 + (lane & 15);
    unsigned short lo = (row0     < Ksrc) ? f2bf_bits(W[(size_t)row0       * FF + col]) : (unsigned short)0;
    unsigned short hi = (row0 + 1 < Ksrc) ? f2bf_bits(W[(size_t)(row0 + 1) * FF + col]) : (unsigned short)0;
    out[(((size_t)nt * KT + (ktBegin + ktl)) * 32 + lane) * 8 + i] = ((unsigned int)hi << 16) | lo;
}

// ---------------- generic WMMA GEMM: C[M,128] = act(A_bf16[M,Kp] @ Bpacked) ----------------
__global__ void gemm_wmma(const unsigned short* __restrict__ A, int Kp,
                          const uint4* __restrict__ Bp,
                          float* __restrict__ Cf, unsigned short* __restrict__ Cb,
                          int M, int doRelu) {
    int lane = threadIdx.x & 31;
    int wid  = blockIdx.x * (blockDim.x >> 5) + (threadIdx.x >> 5);
    int m0 = wid * 16;
    if (m0 >= M) return;
    int r15 = lane & 15, half = lane >> 4;
    int m = m0 + r15;
    int KT = Kp >> 5;
    for (int nt = 0; nt < 8; ++nt) {
        v8f acc = {};
        for (int kt = 0; kt < KT; ++kt) {
            ABu a;
            const unsigned short* ap = A + (size_t)m * Kp + kt * 32 + half * 8;
            a.q[0] = *(const uint4*)ap;
            a.q[1] = *(const uint4*)(ap + 16);
            ABu b;
            const uint4* bp = Bp + ((size_t)(nt * KT + kt) * 32 + lane) * 2;
            b.q[0] = bp[0]; b.q[1] = bp[1];
            acc = wmma_bf16(a, b, acc);
        }
        int col = nt * 16 + r15;
        for (int r = 0; r < 8; ++r) {
            int row = m0 + half * 8 + r;
            float v = acc[r];
            if (doRelu) v = fmaxf(v, 0.f);
            if (Cf) Cf[(size_t)row * FF + col] = v;
            if (Cb) Cb[(size_t)row * FF + col] = f2bf_bits(v);
        }
    }
}

// ---------------- fused CGConv edge kernel ----------------
// One wave per 16-edge tile. z@W decomposed into 10 K-chunks of 32:
//   kt 0-3: h_bf16[dst] ; kt 4-7: h_bf16[src] ; kt 8-9: edge_attr (pad 50->64, cvt inline)
// Two gates (wf, ws) share the A fragment. Gate + scatter-add fused.
__global__ void edge_kernel(const unsigned short* __restrict__ hbf,
                            const float* __restrict__ eattr,
                            const int* __restrict__ ei,     // [2,E]: row0 src, row1 dst
                            const uint4* __restrict__ BpF,
                            const uint4* __restrict__ BpS,
                            const float* __restrict__ biasF,
                            const float* __restrict__ biasS,
                            float* __restrict__ agg) {
    int lane = threadIdx.x & 31;
    int wid  = blockIdx.x * (blockDim.x >> 5) + (threadIdx.x >> 5);
    int eb = wid * 16;
    if (eb >= EE) return;
    int r15 = lane & 15, half = lane >> 4;
    int erow = eb + r15;
    const int* srcI = ei;
    const int* dstI = ei + EE;
    int nd = dstI[erow];
    int ns = srcI[erow];
    int drow[8];
#pragma unroll
    for (int r = 0; r < 8; ++r) drow[r] = dstI[eb + half * 8 + r];

    for (int nt = 0; nt < 8; ++nt) {
        v8f accF = {};
        v8f accS = {};
        for (int kt = 0; kt < 10; ++kt) {
            ABu a;
            if (kt < 8) {
                const unsigned short* ap =
                    hbf + (size_t)(kt < 4 ? nd : ns) * FF + (kt & 3) * 32 + half * 8;
                a.q[0] = *(const uint4*)ap;
                a.q[1] = *(const uint4*)(ap + 16);
            } else {
                int kb = (kt - 8) * 32 + half * 8;
                const float* ep = eattr + (size_t)erow * DD;
#pragma unroll
                for (int j = 0; j < 8; ++j) {
                    int k0 = kb + j, k1 = kb + 16 + j;
                    a.h[j]     = (k0 < DD) ? f2bf_bits(ep[k0]) : (unsigned short)0;
                    a.h[8 + j] = (k1 < DD) ? f2bf_bits(ep[k1]) : (unsigned short)0;
                }
            }
            ABu bF, bS;
            const uint4* pf = BpF + ((size_t)(nt * 10 + kt) * 32 + lane) * 2;
            bF.q[0] = pf[0]; bF.q[1] = pf[1];
            const uint4* ps = BpS + ((size_t)(nt * 10 + kt) * 32 + lane) * 2;
            bS.q[0] = ps[0]; bS.q[1] = ps[1];
            accF = wmma_bf16(a, bF, accF);
            accS = wmma_bf16(a, bS, accS);
        }
        int col = nt * 16 + r15;
        float bfv = biasF[col];
        float bsv = biasS[col];
#pragma unroll
        for (int r = 0; r < 8; ++r) {
            float zf = accF[r] + bfv;
            float zs = accS[r] + bsv;
            float sg = 1.f / (1.f + expf(-zf));
            float sp = (zs > 20.f) ? zs : log1pf(expf(zs));
            atomicAdd(agg + (size_t)drow[r] * FF + col, sg * sp);
        }
    }
}

// ---------------- BatchNorm column stats ----------------
__global__ void col_stats(const float* __restrict__ Xm, int Nrows,
                          float* __restrict__ s1, float* __restrict__ s2) {
    int c = threadIdx.x; // 128 threads
    float a = 0.f, b = 0.f;
    for (long long r = blockIdx.x; r < Nrows; r += gridDim.x) {
        float v = Xm[r * FF + c];
        a += v; b += v * v;
    }
    atomicAdd(&s1[c], a);
    atomicAdd(&s2[c], b);
}

__global__ void bn_finalize(const float* __restrict__ s1, const float* __restrict__ s2,
                            const float* __restrict__ g, const float* __restrict__ b,
                            float invN, float* __restrict__ scale, float* __restrict__ shift) {
    int c = threadIdx.x;
    float mu  = s1[c] * invN;
    float var = s2[c] * invN - mu * mu;
    float sc  = g[c] * rsqrtf(var + 1e-5f);
    scale[c] = sc;
    shift[c] = b[c] - mu * sc;
}

// h = relu(h + agg*scale + shift); refresh bf16 mirror; segment_max via uint atomicMax
__global__ void node_update(float* __restrict__ h, unsigned short* __restrict__ hb,
                            const float* __restrict__ agg,
                            const float* __restrict__ scale, const float* __restrict__ shift,
                            const int* __restrict__ batch,
                            unsigned int* __restrict__ smax, long long total) {
    long long i = (long long)blockIdx.x * blockDim.x + threadIdx.x;
    if (i >= total) return;
    int c = (int)(i & (FF - 1));
    long long node = i >> 7;
    float v = h[i] + agg[i] * scale[c] + shift[c];
    v = fmaxf(v, 0.f);
    h[i]  = v;
    hb[i] = f2bf_bits(v);
    atomicMax(&smax[(size_t)batch[node] * FF + c], __float_as_uint(v)); // valid: v >= 0
}

__global__ void gf_acc(float* __restrict__ gf, unsigned int* __restrict__ smax, int total) {
    int i = blockIdx.x * blockDim.x + threadIdx.x;
    if (i >= total) return;
    gf[i] += __uint_as_float(smax[i]);
    smax[i] = 0u;
}

// ---------------- head ----------------
__global__ void head1(const float* __restrict__ gf, const float* __restrict__ w1,
                      const float* __restrict__ b1, float* __restrict__ y) {
    int g = blockIdx.x, c = threadIdx.x; // 128
    float acc = b1[c];
    for (int k = 0; k < FF; ++k) acc += gf[g * FF + k] * w1[k * FF + c];
    y[g * FF + c] = acc;
}

__global__ void head2(const float* __restrict__ y,
                      const float* __restrict__ scale, const float* __restrict__ shift,
                      const float* __restrict__ w2, const float* __restrict__ b2,
                      float* __restrict__ out) {
    __shared__ float red[FF];
    int g = blockIdx.x, c = threadIdx.x;
    float v = fmaxf(y[g * FF + c] * scale[c] + shift[c], 0.f) * w2[c];
    red[c] = v;
    __syncthreads();
    for (int s = 64; s > 0; s >>= 1) {
        if (c < s) red[c] += red[c + s];
        __syncthreads();
    }
    if (c == 0) out[g] = red[0] + b2[0];
}

// ---------------- host orchestration ----------------
extern "C" void kernel_launch(void* const* d_in, const int* in_sizes, int n_in,
                              void* d_out, int out_size, void* d_ws, size_t ws_size,
                              hipStream_t stream) {
    const float* x      = (const float*)d_in[0];
    const float* eattr  = (const float*)d_in[1];
    const float* w_emb0 = (const float*)d_in[2];
    const float* w_emb1 = (const float*)d_in[3];
    const float* wf     = (const float*)d_in[4];
    const float* bfv    = (const float*)d_in[5];
    const float* wsm    = (const float*)d_in[6];
    const float* bsv    = (const float*)d_in[7];
    const float* bn_g   = (const float*)d_in[8];
    const float* bn_b   = (const float*)d_in[9];
    const float* w1     = (const float*)d_in[10];
    const float* b1     = (const float*)d_in[11];
    const float* bng    = (const float*)d_in[12];
    const float* bnb    = (const float*)d_in[13];
    const float* w2     = (const float*)d_in[14];
    const float* b2     = (const float*)d_in[15];
    const int*   ei     = (const int*)d_in[16];
    const int*   batch  = (const int*)d_in[17];
    float*       out    = (float*)d_out;

    // -------- workspace partition --------
    char* base = (char*)d_ws;
    size_t off = 0;
    auto alloc = [&](size_t bytes) -> void* {
        void* p = base + off;
        off = (off + bytes + 255) & ~(size_t)255;
        return p;
    };
    float*          h    = (float*)alloc((size_t)NN * FF * 4);
    float*          agg  = (float*)alloc((size_t)NN * FF * 4);
    unsigned short* hb   = (unsigned short*)alloc((size_t)NN * FF * 2);
    unsigned short* xb   = (unsigned short*)alloc((size_t)NN * 96 * 2);
    unsigned short* tb   = (unsigned short*)alloc((size_t)NN * FF * 2);
    unsigned int*   pE0  = (unsigned int*)alloc((size_t)3 * 8 * 32 * 8 * 4);
    unsigned int*   pE1  = (unsigned int*)alloc((size_t)4 * 8 * 32 * 8 * 4);
    unsigned int*   pF[LL];
    unsigned int*   pS[LL];
    for (int l = 0; l < LL; ++l) pF[l] = (unsigned int*)alloc((size_t)10 * 8 * 32 * 8 * 4);
    for (int l = 0; l < LL; ++l) pS[l] = (unsigned int*)alloc((size_t)10 * 8 * 32 * 8 * 4);
    float* s1    = (float*)alloc(FF * 4);
    float* s2    = (float*)alloc(FF * 4);
    float* scale = (float*)alloc(FF * 4);
    float* shift = (float*)alloc(FF * 4);
    unsigned int* smax = (unsigned int*)alloc((size_t)GG * FF * 4);
    float* gf    = (float*)alloc((size_t)GG * FF * 4);
    float* y     = (float*)alloc((size_t)GG * FF * 4);

    const int ZT = 256;
    auto zero = [&](void* p, long long n_u32) {
        int blks = (int)((n_u32 + ZT - 1) / ZT);
        zero_u32<<<blks, ZT, 0, stream>>>((unsigned int*)p, n_u32);
    };

    // -------- weight packing (cheap; runs every call for determinism) --------
    {
        long long tot = (long long)NN * 96;
        cvt_pad_bf16<<<(int)((tot + 255) / 256), 256, 0, stream>>>(x, F_IN, 96, xb, tot);
    }
    packB<<<(3 * 8 * 32 * 8 + 255) / 256, 256, 0, stream>>>(w_emb0, F_IN, 0, 3, 3, pE0);
    packB<<<(4 * 8 * 32 * 8 + 255) / 256, 256, 0, stream>>>(w_emb1, FF, 0, 4, 4, pE1);
    for (int l = 0; l < LL; ++l) {
        const float* wfl = wf + (size_t)l * (2 * FF + DD) * FF;
        const float* wsl = wsm + (size_t)l * (2 * FF + DD) * FF;
        int pb = (4 * 8 * 32 * 8 + 255) / 256;
        int pe = (2 * 8 * 32 * 8 + 255) / 256;
        packB<<<pb, 256, 0, stream>>>(wfl,                 FF, 0, 4, 10, pF[l]); // dst block
        packB<<<pb, 256, 0, stream>>>(wfl + 128 * FF,      FF, 4, 4, 10, pF[l]); // src block
        packB<<<pe, 256, 0, stream>>>(wfl + 256 * FF,      DD, 8, 2, 10, pF[l]); // edge block
        packB<<<pb, 256, 0, stream>>>(wsl,                 FF, 0, 4, 10, pS[l]);
        packB<<<pb, 256, 0, stream>>>(wsl + 128 * FF,      FF, 4, 4, 10, pS[l]);
        packB<<<pe, 256, 0, stream>>>(wsl + 256 * FF,      DD, 8, 2, 10, pS[l]);
    }

    // -------- embedding: h = relu(x @ w_emb0) @ w_emb1 --------
    {
        int waves = NN / 16;                 // 3125
        int blks = (waves + 7) / 8;
        gemm_wmma<<<blks, 256, 0, stream>>>(xb, 96, (const uint4*)pE0,
                                            nullptr, tb, NN, 1);
        gemm_wmma<<<blks, 256, 0, stream>>>(tb, 128, (const uint4*)pE1,
                                            h, hb, NN, 0);
    }

    zero(gf, (long long)GG * FF);
    zero(smax, (long long)GG * FF);

    // -------- CGConv layers --------
    for (int l = 0; l < LL; ++l) {
        zero(agg, (long long)NN * FF);
        zero(s1, FF);
        zero(s2, FF);

        int eblks = (EE / 16 + 7) / 8;       // 8 waves per block
        edge_kernel<<<eblks, 256, 0, stream>>>(hb, eattr, ei,
                                               (const uint4*)pF[l], (const uint4*)pS[l],
                                               bfv + (size_t)l * FF, bsv + (size_t)l * FF,
                                               agg);

        col_stats<<<256, FF, 0, stream>>>(agg, NN, s1, s2);
        bn_finalize<<<1, FF, 0, stream>>>(s1, s2, bn_g + (size_t)l * FF, bn_b + (size_t)l * FF,
                                          1.0f / (float)NN, scale, shift);

        long long tot = (long long)NN * FF;
        node_update<<<(int)((tot + 255) / 256), 256, 0, stream>>>(h, hb, agg, scale, shift,
                                                                  batch, smax, tot);
        gf_acc<<<(GG * FF + 255) / 256, 256, 0, stream>>>(gf, smax, GG * FF);
    }

    // -------- head --------
    head1<<<GG, FF, 0, stream>>>(gf, w1, b1, y);
    zero(s1, FF);
    zero(s2, FF);
    col_stats<<<64, FF, 0, stream>>>(y, GG, s1, s2);
    bn_finalize<<<1, FF, 0, stream>>>(s1, s2, bng, bnb, 1.0f / (float)GG, scale, shift);
    head2<<<GG, FF, 0, stream>>>(y, scale, shift, w2, b2, out);
}
